// CrossNonLocalBlock_5196910428637
// MI455X (gfx1250) — compile-verified, matrix operand order checked by reading, not run on hardware
//
#include <hip/hip_runtime.h>
#include <hip/hip_bf16.h>

#define BATCH 8
#define CS    256
#define CT    256
#define CI    128
#define NPIX  4096          // 64*64
#define BN_EPS 1e-5f

typedef _Float16 v16h  __attribute__((ext_vector_type(16)));
typedef _Float16 half8 __attribute__((ext_vector_type(8)));
typedef float    v8f   __attribute__((ext_vector_type(8)));

union HVec { v16h v; half8 h[2]; };

static __device__ __forceinline__ v8f wmma_f16(v16h a, v16h b, v8f c) {
  // D = A(16x32 f16) * B(32x16 f16) + C(16x16 f32)
  return __builtin_amdgcn_wmma_f32_16x16x32_f16(false, a, false, b, (short)0, c,
                                                false, false);
}

static __device__ __forceinline__ v8f vzero8() {
  v8f z = {0.f, 0.f, 0.f, 0.f, 0.f, 0.f, 0.f, 0.f};
  return z;
}

// ---------------------------------------------------------------------------
// Kernel 1: 1x1-conv projections -> f16 workspace tensors in WMMA-friendly
// layouts.  theta: [b][n][c], phiT: [b][m][c], gT: [b][c][n].
// grid = 3 * BATCH*CI*(NPIX/256), block = 256
// ---------------------------------------------------------------------------
__global__ __launch_bounds__(256) void proj_kernel(
    const float* __restrict__ x, const float* __restrict__ l,
    const float* __restrict__ gw, const float* __restrict__ gb,
    const float* __restrict__ tw, const float* __restrict__ tb,
    const float* __restrict__ pw, const float* __restrict__ pb,
    _Float16* __restrict__ theta_h, _Float16* __restrict__ phi_h,
    _Float16* __restrict__ gT_h) {
  const int blocksPer = BATCH * CI * (NPIX / 256);   // 16384
  int pid = blockIdx.x / blocksPer;                  // 0=theta 1=g 2=phi
  int rem = blockIdx.x % blocksPer;
  int nblk = rem & 15;
  int bc   = rem >> 4;
  int b = bc / CI, c = bc % CI;
  int n = nblk * 256 + threadIdx.x;

  const float* in;
  const float* w;
  float bias;
  if (pid == 0)      { in = x; w = tw; bias = tb[c]; }
  else if (pid == 1) { in = x; w = gw; bias = gb[c]; }
  else               { in = l; w = pw; bias = pb[c]; }

  const float* xp = in + (size_t)b * CS * NPIX + n;
  const float* wp = w + (size_t)c * CS;     // uniform per block -> s_loads
  float acc = bias;
#pragma unroll 4
  for (int cs = 0; cs < CS; ++cs)
    acc += xp[(size_t)cs * NPIX] * wp[cs];

  _Float16 hv = (_Float16)acc;
  if (pid == 0)      theta_h[((size_t)b * NPIX + n) * CI + c] = hv;
  else if (pid == 1) gT_h[((size_t)b * CI + c) * NPIX + n]   = hv;
  else               phi_h[((size_t)b * NPIX + n) * CI + c]  = hv;
}

// ---------------------------------------------------------------------------
// Kernel 2 (pass 1): per-row logsumexp of S = theta * phi^T.
// One wave (32 threads) owns a 16-row strip of n; processes 64 m-columns
// (4 WMMA tiles) per reduction round so cross-lane butterflies are amortized
// 4x against the 16 WMMAs.  Writes shift[n] = rowmax + log(rowsum).
// grid = BATCH*(NPIX/16), block = 32
// ---------------------------------------------------------------------------
__global__ __launch_bounds__(32) void attn_rowstats_kernel(
    const _Float16* __restrict__ theta_h, const _Float16* __restrict__ phi_h,
    float* __restrict__ shift) {
  const int lane = threadIdx.x;
  const int li = lane & 15, grp = lane >> 4, cb = grp * 8;
  const int b  = blockIdx.x >> 8;          // 256 row-tiles per batch
  const int t  = blockIdx.x & 255;
  const int n0 = t * 16;

  const _Float16* th = theta_h + (size_t)b * NPIX * CI;
  const _Float16* ph = phi_h + (size_t)b * NPIX * CI;

  // A tiles (theta rows n0..n0+15, K = Ci in 4 chunks of 32) stay resident.
  HVec a[4];
#pragma unroll
  for (int k = 0; k < 4; ++k) {
    size_t base = (size_t)(n0 + li) * CI + k * 32 + cb;
    a[k].h[0] = *(const half8*)(th + base);
    a[k].h[1] = *(const half8*)(th + base + 16);
  }

  float rm[8], rs[8];
#pragma unroll
  for (int v = 0; v < 8; ++v) { rm[v] = -1e30f; rs[v] = 0.f; }

  for (int mt = 0; mt < NPIX / 64; ++mt) {
    const int m0 = mt * 64;
    v8f s[4];
#pragma unroll
    for (int j = 0; j < 4; ++j) {
      s[j] = vzero8();
#pragma unroll
      for (int k = 0; k < 4; ++k) {
        HVec bt;
        size_t base = (size_t)(m0 + j * 16 + li) * CI + k * 32 + cb;
        bt.h[0] = *(const half8*)(ph + base);
        bt.h[1] = *(const half8*)(ph + base + 16);
        s[j] = wmma_f16(a[k].v, bt.v, s[j]);
      }
    }
    // online logsumexp: row r = v + 8*grp; 64 columns = 4 tile elems x 16 lanes
#pragma unroll
    for (int v = 0; v < 8; ++v) {
      float mx = fmaxf(fmaxf(s[0][v], s[1][v]), fmaxf(s[2][v], s[3][v]));
#pragma unroll
      for (int msk = 1; msk < 16; msk <<= 1)
        mx = fmaxf(mx, __shfl_xor(mx, msk, 32));
      float nm = fmaxf(rm[v], mx);
      float p = __expf(s[0][v] - nm) + __expf(s[1][v] - nm) +
                __expf(s[2][v] - nm) + __expf(s[3][v] - nm);
#pragma unroll
      for (int msk = 1; msk < 16; msk <<= 1)
        p += __shfl_xor(p, msk, 32);
      rs[v] = rs[v] * __expf(rm[v] - nm) + p;
      rm[v] = nm;
    }
  }

  if (li == 0) {
#pragma unroll
    for (int v = 0; v < 8; ++v) {
      int r = n0 + v + 8 * grp;
      shift[(size_t)b * NPIX + r] = rm[v] + __logf(rs[v]);
    }
  }
}

// ---------------------------------------------------------------------------
// Kernel 3 (pass 2): y[m,c] = sum_n P[n,m] * g[n,c], P = exp(S - shift[n]).
// One wave owns a 16-column m-tile, loops over n in steps of 32:
//  - recompute two 16x16 S tiles with WMMA (B=phi tiles resident),
//  - P = exp(S - logsumexp), transpose P into LDS as f16,
//  - A2 = P^T (16m x 32n) from LDS, B2 = g (32n x 16c) from gT -> 8 WMMAs.
// grid = BATCH*(NPIX/16), block = 32
// ---------------------------------------------------------------------------
__global__ __launch_bounds__(32) void attn_apply_kernel(
    const _Float16* __restrict__ theta_h, const _Float16* __restrict__ phi_h,
    const _Float16* __restrict__ gT_h, const float* __restrict__ shift,
    float* __restrict__ yT) {
  __shared__ _Float16 lds_p[16 * 32];   // [m][n] tile, n contiguous

  const int lane = threadIdx.x;
  const int li = lane & 15, grp = lane >> 4, cb = grp * 8;
  const int b  = blockIdx.x >> 8;
  const int t  = blockIdx.x & 255;
  const int m0 = t * 16;

  const _Float16* th = theta_h + (size_t)b * NPIX * CI;
  const _Float16* ph = phi_h + (size_t)b * NPIX * CI;
  const _Float16* gt = gT_h + (size_t)b * CI * NPIX;
  const size_t bN = (size_t)b * NPIX;

  // phi B-tiles for this wave's 16 m columns stay resident (K = Ci).
  HVec bphi[4];
#pragma unroll
  for (int k = 0; k < 4; ++k) {
    size_t base = (size_t)(m0 + li) * CI + k * 32 + cb;
    bphi[k].h[0] = *(const half8*)(ph + base);
    bphi[k].h[1] = *(const half8*)(ph + base + 16);
  }

  v8f acc[8];
#pragma unroll
  for (int ct = 0; ct < 8; ++ct) acc[ct] = vzero8();

  for (int nt = 0; nt < NPIX / 32; ++nt) {
#pragma unroll
    for (int hh = 0; hh < 2; ++hh) {
      const int n0 = nt * 32 + hh * 16;
      v8f s = vzero8();
#pragma unroll
      for (int k = 0; k < 4; ++k) {
        HVec at;
        size_t base = (size_t)(n0 + li) * CI + k * 32 + cb;
        at.h[0] = *(const half8*)(th + base);
        at.h[1] = *(const half8*)(th + base + 16);
        s = wmma_f16(at.v, bphi[k].v, s);
      }
      // every lane fetches shift[n0 + li]; rows 0..15 live in lanes 0..15
      float tshift = shift[bN + n0 + li];
#pragma unroll
      for (int v = 0; v < 8; ++v) {
        int r = v + 8 * grp;                    // local row of this element
        float sh = __shfl(tshift, r, 32);
        float p = __expf(s[v] - sh);
        lds_p[li * 32 + hh * 16 + r] = (_Float16)p;   // transposed store
      }
    }
    asm volatile("s_wait_dscnt 0" ::: "memory");

    // A2 = P^T : row m = li, K = n chunks (cb..cb+7, 16+cb..16+cb+7)
    HVec a2;
    a2.h[0] = *(const half8*)(&lds_p[li * 32 + cb]);
    a2.h[1] = *(const half8*)(&lds_p[li * 32 + 16 + cb]);

    const int nb = nt * 32;
#pragma unroll
    for (int ct = 0; ct < 8; ++ct) {
      HVec b2;   // B2 element (K=n, N=c): gT row c gives contiguous n
      size_t base = (size_t)(ct * 16 + li) * NPIX + nb;
      b2.h[0] = *(const half8*)(gt + base + cb);
      b2.h[1] = *(const half8*)(gt + base + 16 + cb);
      acc[ct] = wmma_f16(a2.v, b2.v, acc[ct]);
    }
  }

  // write y as [b][c][n] f32 so the W-conv streams coalesced in n
#pragma unroll
  for (int ct = 0; ct < 8; ++ct) {
#pragma unroll
    for (int v = 0; v < 8; ++v) {
      int c = ct * 16 + li;
      int m = m0 + v + 8 * grp;
      yT[((size_t)b * CI + c) * NPIX + m] = acc[ct][v];
    }
  }
}

// ---------------------------------------------------------------------------
// Kernel 4: wy = W*y + b, plus deterministic per-block BN partial sums.
// grid = BATCH*CT*(NPIX/256), block = 256
// ---------------------------------------------------------------------------
__global__ __launch_bounds__(256) void wconv_kernel(
    const float* __restrict__ yT, const float* __restrict__ ww,
    const float* __restrict__ wb, float* __restrict__ wy,
    float* __restrict__ part1, float* __restrict__ part2) {
  int rem = blockIdx.x;
  int nblk = rem & 15;
  int bc = rem >> 4;
  int b = bc / CT, ct = bc % CT;
  int n = nblk * 256 + threadIdx.x;

  const float* yp = yT + (size_t)b * CI * NPIX + n;
  const float* wp = ww + (size_t)ct * CI;     // uniform -> scalar loads
  float acc = wb[ct];
#pragma unroll 4
  for (int c = 0; c < CI; ++c)
    acc += yp[(size_t)c * NPIX] * wp[c];
  wy[((size_t)b * CT + ct) * NPIX + n] = acc;

  __shared__ float s1[256], s2[256];
  s1[threadIdx.x] = acc;
  s2[threadIdx.x] = acc * acc;
  __syncthreads();
  for (int stride = 128; stride > 0; stride >>= 1) {
    if (threadIdx.x < stride) {
      s1[threadIdx.x] += s1[threadIdx.x + stride];
      s2[threadIdx.x] += s2[threadIdx.x + stride];
    }
    __syncthreads();
  }
  if (threadIdx.x == 0) {
    int slot = b * 16 + nblk;                 // 128 partials per channel
    part1[(size_t)ct * 128 + slot] = s1[0];
    part2[(size_t)ct * 128 + slot] = s2[0];
  }
}

// grid = CT, block = 128 : deterministic channel reduction
__global__ __launch_bounds__(128) void stats_kernel(
    const float* __restrict__ part1, const float* __restrict__ part2,
    float* __restrict__ stats) {
  int ct = blockIdx.x;
  __shared__ float s1[128], s2[128];
  s1[threadIdx.x] = part1[(size_t)ct * 128 + threadIdx.x];
  s2[threadIdx.x] = part2[(size_t)ct * 128 + threadIdx.x];
  __syncthreads();
  for (int stride = 64; stride > 0; stride >>= 1) {
    if (threadIdx.x < stride) {
      s1[threadIdx.x] += s1[threadIdx.x + stride];
      s2[threadIdx.x] += s2[threadIdx.x + stride];
    }
    __syncthreads();
  }
  if (threadIdx.x == 0) {
    stats[ct]      = s1[0];
    stats[CT + ct] = s2[0];
  }
}

// ---------------------------------------------------------------------------
// Kernel 5: BN (batch stats) + affine + residual.  grid = B*CT*NPIX/256
// ---------------------------------------------------------------------------
__global__ __launch_bounds__(256) void finalize_kernel(
    const float* __restrict__ wy, const float* __restrict__ stats,
    const float* __restrict__ gamma, const float* __restrict__ beta,
    const float* __restrict__ l, float* __restrict__ out) {
  size_t id = (size_t)blockIdx.x * 256 + threadIdx.x;
  int ct = (int)((id / NPIX) % CT);
  const float invBN = 1.f / (float)(BATCH * NPIX);
  float mean = stats[ct] * invBN;
  float var = stats[CT + ct] * invBN - mean * mean;
  float v = (wy[id] - mean) * rsqrtf(var + BN_EPS) * gamma[ct] + beta[ct] + l[id];
  out[id] = v;
}

// ---------------------------------------------------------------------------
extern "C" void kernel_launch(void* const* d_in, const int* in_sizes, int n_in,
                              void* d_out, int out_size, void* d_ws,
                              size_t ws_size, hipStream_t stream) {
  const float* x   = (const float*)d_in[0];
  const float* l   = (const float*)d_in[1];
  const float* g_w = (const float*)d_in[2];
  const float* g_b = (const float*)d_in[3];
  const float* t_w = (const float*)d_in[4];
  const float* t_b = (const float*)d_in[5];
  const float* p_w = (const float*)d_in[6];
  const float* p_b = (const float*)d_in[7];
  const float* w_w = (const float*)d_in[8];
  const float* w_b = (const float*)d_in[9];
  const float* bn_g = (const float*)d_in[10];
  const float* bn_b = (const float*)d_in[11];
  float* out = (float*)d_out;

  // workspace carve-up
  char* p = (char*)d_ws;
  const size_t SZ_H = (size_t)BATCH * NPIX * CI * sizeof(_Float16); // 8 MB
  _Float16* theta_h = (_Float16*)(p);
  _Float16* phi_h   = (_Float16*)(p + SZ_H);
  _Float16* gT_h    = (_Float16*)(p + 2 * SZ_H);
  float* shift  = (float*)(p + 3 * SZ_H);
  float* yT     = shift + (size_t)BATCH * NPIX;
  float* wy     = yT + (size_t)BATCH * CI * NPIX;
  float* part1  = wy + (size_t)BATCH * CT * NPIX;
  float* part2  = part1 + (size_t)CT * 128;
  float* stats  = part2 + (size_t)CT * 128;

  proj_kernel<<<3 * BATCH * CI * (NPIX / 256), 256, 0, stream>>>(
      x, l, g_w, g_b, t_w, t_b, p_w, p_b, theta_h, phi_h, gT_h);

  attn_rowstats_kernel<<<BATCH * (NPIX / 16), 32, 0, stream>>>(
      theta_h, phi_h, shift);

  attn_apply_kernel<<<BATCH * (NPIX / 16), 32, 0, stream>>>(
      theta_h, phi_h, gT_h, shift, yT);

  wconv_kernel<<<BATCH * CT * (NPIX / 256), 256, 0, stream>>>(
      yT, w_w, w_b, wy, part1, part2);

  stats_kernel<<<CT, 128, 0, stream>>>(part1, part2, stats);

  finalize_kernel<<<(BATCH * CT * NPIX) / 256, 256, 0, stream>>>(
      wy, stats, bn_g, bn_b, l, out);
}
